// AttentionBased_32976758899050
// MI455X (gfx1250) — matrix operand order repaired and checked by reading, not compiled, not run on hardware
//
#include <hip/hip_runtime.h>
#include <hip/hip_bf16.h>
#include <stdint.h>

typedef __attribute__((ext_vector_type(16))) _Float16 v16h;
typedef __attribute__((ext_vector_type(8)))  _Float16 v8h;
typedef __attribute__((ext_vector_type(8)))  float    v8f;

#define NB 4  // batch

// All internal activation tensors are f16 NHWC: act[b][y][x][c], c fastest.

// ---------------- conv1: 3->64 direct, f32 NCHW in -> f16 NHWC out ----------
__global__ void k_conv_first(const float* __restrict__ x, const float* __restrict__ w,
                             const float* __restrict__ bias, _Float16* __restrict__ out,
                             int H, int W) {
  const int npix = H * W;
  long long idx   = (long long)blockIdx.x * blockDim.x + threadIdx.x;
  long long total = (long long)NB * npix * 64;
  if (idx >= total) return;
  int co = (int)(idx & 63);
  int p  = (int)((idx >> 6) % npix);
  int b  = (int)(idx / ((long long)npix * 64));
  int y = p / W, xp = p - y * W;
  float acc = bias[co];
#pragma unroll
  for (int c = 0; c < 3; ++c) {
#pragma unroll
    for (int t = 0; t < 9; ++t) {
      int yy = y + t / 3 - 1, xx = xp + t % 3 - 1;
      if (yy >= 0 && yy < H && xx >= 0 && xx < W)
        acc += x[((long long)(b * 3 + c) * H + yy) * W + xx] * w[(co * 3 + c) * 9 + t];
    }
  }
  out[idx] = (_Float16)fmaxf(acc, 0.0f);  // idx == ((b*npix+p)*64 + co)
}

// ---------------- generic 3x3 conv, implicit GEMM on v_wmma_f32_16x16x32_f16 --
// One wave -> 16(Cout) x 16(pixel) tile; 8 waves/block share a 16xK f16 weight
// panel in LDS (K = Cin*9, tap-major). Tap-outer / channel-chunk-inner loop:
// per tap the address math + bounds test is invariant; the unrolled inner loop
// is pure ds_load_b128 + global_load_b128 + v_wmma, with branchless zero-fill
// (clamped address + select) so EXEC stays full.
template<int CIN>
__global__ void k_conv_wmma(const _Float16* __restrict__ act, const float* __restrict__ wgt,
                            const float* __restrict__ bias, _Float16* __restrict__ out,
                            int H, int W, int Cout, int fuse_bn,
                            const float* __restrict__ bng, const float* __restrict__ bnb,
                            const float* __restrict__ bnm, const float* __restrict__ bnv) {
  constexpr int K    = CIN * 9;
  constexpr int Kpad = K + 8;                 // +16B pad: distinct LDS banks for A reads
  constexpr int CSH  = (CIN == 128) ? 7 : 6;
  __shared__ __align__(16) _Float16 ldsW[16 * Kpad];

  const int tid   = threadIdx.x;
  const int cout0 = blockIdx.y * 16;
  const int bz    = blockIdx.z;
  const int npix  = H * W;

  // stage f32 OIHW weights -> f16 LDS panel [16 x Kpad], tap-major K
  for (int i = tid; i < 16 * K; i += 256) {
    int m = i / K, k = i - m * K;
    int t = k >> CSH;
    int c = k & (CIN - 1);
    ldsW[m * Kpad + k] = (_Float16)wgt[((cout0 + m) * CIN + c) * 9 + t];
  }
  __syncthreads();

  const int  lane = tid & 31;
  const int  wv   = tid >> 5;
  const int  pix0 = blockIdx.x * 128 + wv * 16;
  const int  n    = lane & 15;
  const int  p    = pix0 + n;
  const bool pv   = p < npix;
  const int  pc   = pv ? p : (npix - 1);
  const int  y    = pc / W;
  const int  x    = pc - y * W;
  const int  offA = (lane < 16) ? 0 : 8;    // A layout: hi half-wave holds K+8
  const int  offB = (lane < 16) ? 0 : 8;    // B layout: hi half-wave holds K+8
  const _Float16* actb = act + (long long)bz * npix * CIN;
  const _Float16* arow = &ldsW[(lane & 15) * Kpad + offA];

  v8f acc = {0.f, 0.f, 0.f, 0.f, 0.f, 0.f, 0.f, 0.f};

  for (int t = 0; t < 9; ++t) {
    const int  yy = y + t / 3 - 1;
    const int  xx = x + t % 3 - 1;
    const bool ok = pv & (yy >= 0) & (yy < H) & (xx >= 0) & (xx < W);
    const int  yc = ok ? yy : 0;
    const int  xc = ok ? xx : 0;
    const _Float16* bp = actb + ((long long)yc * W + xc) * CIN + offB;  // always valid
    const _Float16* ap = arow + t * CIN;
#pragma unroll
    for (int cc = 0; cc < CIN; cc += 32) {
      // A fragment (weights) from LDS
      v8h lo = *(const v8h*)(ap + cc);
      v8h hi = *(const v8h*)(ap + cc + 16);
      v16h afrag;
#pragma unroll
      for (int i = 0; i < 8; ++i) { afrag[i] = lo[i]; afrag[i + 8] = hi[i]; }

      // B fragment: two 8-channel runs at one pixel (k and k+16)
      v8h b0 = *(const v8h*)(bp + cc);
      v8h b1 = *(const v8h*)(bp + cc + 16);
      v16h bfrag;
#pragma unroll
      for (int h = 0; h < 8; ++h) {
        bfrag[h]     = ok ? b0[h] : (_Float16)0.f;
        bfrag[h + 8] = ok ? b1[h] : (_Float16)0.f;
      }

      acc = __builtin_amdgcn_wmma_f32_16x16x32_f16(false, afrag, false, bfrag,
                                                   (short)0, acc, false, false);
    }
  }

  if (pv) {
    v8h ov;
#pragma unroll
    for (int r = 0; r < 8; ++r) {
      int   m  = (lane < 16) ? r : (8 + r);   // C/D layout: hi half-wave holds M+8
      int   co = cout0 + m;
      float v  = acc[r] + bias[co];
      v = fmaxf(v, 0.0f);
      if (fuse_bn) {
        float s = bng[co] * rsqrtf(bnv[co] + 1e-5f);
        v = (v - bnm[co]) * s + bnb[co];
      }
      ov[r] = (_Float16)v;
    }
    // 8 consecutive output channels -> single 16B store
    _Float16* op = out + ((long long)bz * npix + p) * Cout + cout0 + ((lane < 16) ? 0 : 8);
    *(v8h*)op = ov;
  }
}

// ---------------- 2x2 max pool (NHWC) --------------------------------------
__global__ void k_maxpool2(const _Float16* __restrict__ in, _Float16* __restrict__ out,
                           int C, int H, int W) {
  int Ho = H >> 1, Wo = W >> 1;
  long long total = (long long)NB * Ho * Wo * C;
  long long idx   = (long long)blockIdx.x * blockDim.x + threadIdx.x;
  if (idx >= total) return;
  int c  = (int)(idx % C);
  int xo = (int)((idx / C) % Wo);
  int yo = (int)((idx / ((long long)C * Wo)) % Ho);
  int b  = (int)(idx / ((long long)C * Wo * Ho));
  const _Float16* s = in + (((long long)b * H + 2 * yo) * W + 2 * xo) * C + c;
  const long long rs = (long long)W * C;
  float a0 = (float)s[0], a1 = (float)s[C], a2 = (float)s[rs], a3 = (float)s[rs + C];
  out[idx] = (_Float16)fmaxf(fmaxf(a0, a1), fmaxf(a2, a3));
}

// ---------------- bilinear 2x upsample, align_corners=True (NHWC) ----------
__global__ void k_up2x(const _Float16* __restrict__ in, _Float16* __restrict__ out,
                       int C, int H, int W) {
  int Ho = 2 * H, Wo = 2 * W;
  long long total = (long long)NB * Ho * Wo * C;
  long long idx   = (long long)blockIdx.x * blockDim.x + threadIdx.x;
  if (idx >= total) return;
  int c  = (int)(idx % C);
  int xo = (int)((idx / C) % Wo);
  int yo = (int)((idx / ((long long)C * Wo)) % Ho);
  int b  = (int)(idx / ((long long)C * Wo * Ho));
  float sy = (float)(H - 1) / (float)(Ho - 1);
  float sx = (float)(W - 1) / (float)(Wo - 1);
  float fy = yo * sy, fx = xo * sx;
  int y0 = (int)fy, x0 = (int)fx;
  int y1 = (y0 + 1 < H) ? y0 + 1 : H - 1;
  int x1 = (x0 + 1 < W) ? x0 + 1 : W - 1;
  float wy = fy - y0, wx = fx - x0;
  const _Float16* s = in + (long long)b * H * W * C + c;
  float v00 = (float)s[((long long)y0 * W + x0) * C], v01 = (float)s[((long long)y0 * W + x1) * C];
  float v10 = (float)s[((long long)y1 * W + x0) * C], v11 = (float)s[((long long)y1 * W + x1) * C];
  float r0 = v00 * (1.f - wx) + v01 * wx;
  float r1 = v10 * (1.f - wx) + v11 * wx;
  out[idx] = (_Float16)(r0 * (1.f - wy) + r1 * wy);
}

// ---------------- block attention: w[m,n] = <blk(m,n), sum_all_blks>/676 ----
__global__ void k_zero16(float* __restrict__ w) {
  if (threadIdx.x < 16) w[threadIdx.x] = 0.f;
}

// T[b][h][w][c] (f32, c fastest), x is NHWC [b][104][104][128]
__global__ void k_attn_tsum(const _Float16* __restrict__ x, float* __restrict__ T) {
  int idx = blockIdx.x * 256 + threadIdx.x;
  const int total = NB * 26 * 26 * 128;
  if (idx >= total) return;
  int c = idx & 127;
  int w = (idx >> 7) % 26;
  int h = (idx / (128 * 26)) % 26;
  int b = idx / (128 * 676);
  const _Float16* base = x + (long long)b * 104 * 104 * 128 + c;
  float s = 0.f;
#pragma unroll
  for (int m = 0; m < 4; ++m)
#pragma unroll
    for (int n = 0; n < 4; ++n)
      s += (float)base[((long long)(m * 26 + h) * 104 + n * 26 + w) * 128];
  T[idx] = s;
}

__global__ void k_attn_dot(const _Float16* __restrict__ x, const float* __restrict__ T,
                           float* __restrict__ wsum) {
  __shared__ float sb[16];
  int tid = threadIdx.x;
  if (tid < 16) sb[tid] = 0.f;
  __syncthreads();
  int idx = blockIdx.x * 256 + tid;
  const int total = NB * 26 * 26 * 128;
  if (idx < total) {
    int c = idx & 127;
    int w = (idx >> 7) % 26;
    int h = (idx / (128 * 26)) % 26;
    int b = idx / (128 * 676);
    const _Float16* base = x + (long long)b * 104 * 104 * 128 + c;
    float tv = T[idx];
#pragma unroll
    for (int m = 0; m < 4; ++m)
#pragma unroll
      for (int n = 0; n < 4; ++n)
        atomicAdd(&sb[m * 4 + n],
                  tv * (float)base[((long long)(m * 26 + h) * 104 + n * 26 + w) * 128]);
  }
  __syncthreads();
  if (tid < 16) atomicAdd(&wsum[tid], sb[tid]);
}

__global__ void k_attn_scale(_Float16* __restrict__ x, const float* __restrict__ wsum) {
  long long idx   = (long long)blockIdx.x * blockDim.x + threadIdx.x;
  long long total = (long long)NB * 104 * 104 * 128;
  if (idx >= total) return;
  int X = (int)((idx >> 7) % 104);
  int Y = (int)((idx / (128 * 104)) % 104);
  float s = wsum[(Y / 26) * 4 + (X / 26)] * (1.0f / 676.0f);
  x[idx] = (_Float16)((float)x[idx] * s);
}

// ---------------- last conv: 64->2 direct + sigmoid, NHWC in, NCHW f32 out --
// Weights staged to LDS with gfx1250 async copy (global_load_async_to_lds_b128).
__global__ void k_conv_last(const _Float16* __restrict__ act, const float* __restrict__ w,
                            const float* __restrict__ bias, float* __restrict__ out,
                            int H, int W) {
  __shared__ __align__(16) float wl[2 * 64 * 9];  // 4608 B = 288 x 16B chunks
  int tid = threadIdx.x;
  for (int i = tid; i < 288; i += 256) {
    unsigned           lws = (unsigned)(uintptr_t)&wl[4 * i];      // LDS byte address
    unsigned long long gsv = (unsigned long long)(uintptr_t)(w + 4 * i);
    asm volatile("global_load_async_to_lds_b128 %0, %1, off"
                 :: "v"(lws), "v"(gsv) : "memory");
  }
  asm volatile("s_wait_asynccnt 0" ::: "memory");
  __syncthreads();

  const int npix = H * W;
  long long idx   = (long long)blockIdx.x * 256 + tid;
  long long total = (long long)NB * npix;
  if (idx < total) {
    int p = (int)(idx % npix), b = (int)(idx / npix);
    int y = p / W, xp = p - y * W;
    float a0 = bias[0], a1 = bias[1];
    const _Float16* ab = act + (long long)b * npix * 64;
#pragma unroll
    for (int t = 0; t < 9; ++t) {
      int yy = y + t / 3 - 1, xx = xp + t % 3 - 1;
      if (yy >= 0 && yy < H && xx >= 0 && xx < W) {
        const _Float16* ap = ab + ((long long)yy * W + xx) * 64;
        for (int c = 0; c < 64; ++c) {
          float a = (float)ap[c];
          a0 += a * wl[c * 9 + t];
          a1 += a * wl[576 + c * 9 + t];
        }
      }
    }
    out[(long long)(b * 2 + 0) * npix + p] = 1.0f / (1.0f + __expf(-a0));
    out[(long long)(b * 2 + 1) * npix + p] = 1.0f / (1.0f + __expf(-a1));
  }
}

// ---------------------------------------------------------------------------
extern "C" void kernel_launch(void* const* d_in, const int* in_sizes, int n_in,
                              void* d_out, int out_size, void* d_ws, size_t ws_size,
                              hipStream_t stream) {
  (void)in_sizes; (void)n_in; (void)out_size; (void)ws_size;
  const float* x    = (const float*)d_in[0];
  const float* w1   = (const float*)d_in[1];  const float* b1  = (const float*)d_in[2];
  const float* w2   = (const float*)d_in[3];  const float* b2  = (const float*)d_in[4];
  const float* w3   = (const float*)d_in[5];  const float* b3  = (const float*)d_in[6];
  const float* w4   = (const float*)d_in[7];  const float* b4  = (const float*)d_in[8];
  const float* dw1  = (const float*)d_in[9];  const float* db1 = (const float*)d_in[10];
  const float* dw2  = (const float*)d_in[11]; const float* db2 = (const float*)d_in[12];
  const float* dw3  = (const float*)d_in[13]; const float* db3 = (const float*)d_in[14];
  const float* dw4  = (const float*)d_in[15]; const float* db4 = (const float*)d_in[16];
  const float* dw5  = (const float*)d_in[17]; const float* db5 = (const float*)d_in[18];
  const float* bn2g = (const float*)d_in[19]; const float* bn2b= (const float*)d_in[20];
  const float* bn2m = (const float*)d_in[21]; const float* bn2v= (const float*)d_in[22];
  const float* bn4g = (const float*)d_in[23]; const float* bn4b= (const float*)d_in[24];
  const float* bn4m = (const float*)d_in[25]; const float* bn4v= (const float*)d_in[26];

  // workspace: two f16 NHWC ping-pong activation buffers + attention scratch
  const size_t BUFBYTES = 88604672ULL;               // 4*416*416*64 * 2B
  _Float16* A  = (_Float16*)d_ws;
  _Float16* Bf = (_Float16*)((char*)d_ws + BUFBYTES);
  float*    T    = (float*)((char*)d_ws + 2 * BUFBYTES);                 // 4*26*26*128 f32
  float*    wsum = (float*)((char*)d_ws + 2 * BUFBYTES + 1384448ULL);    // 16 f32
  float*    outp = (float*)d_out;

  // encoder
  {
    long long total = 4LL * 416 * 416 * 64;
    k_conv_first<<<dim3((unsigned)((total + 255) / 256)), 256, 0, stream>>>(x, w1, b1, A, 416, 416);
  }
  k_conv_wmma<64><<<dim3(1352, 4, 4), 256, 0, stream>>>(A, w2, b2, Bf, 416, 416, 64, 0, b2, b2, b2, b2);
  {
    long long total = 4LL * 208 * 208 * 64;
    k_maxpool2<<<dim3((unsigned)((total + 255) / 256)), 256, 0, stream>>>(Bf, A, 64, 416, 416);
  }
  k_conv_wmma<64><<<dim3(338, 8, 4), 256, 0, stream>>>(A, w3, b3, Bf, 208, 208, 128, 0, b3, b3, b3, b3);
  k_conv_wmma<128><<<dim3(338, 8, 4), 256, 0, stream>>>(Bf, w4, b4, A, 208, 208, 128, 0, b4, b4, b4, b4);
  {
    long long total = 4LL * 104 * 104 * 128;
    k_maxpool2<<<dim3((unsigned)((total + 255) / 256)), 256, 0, stream>>>(A, Bf, 128, 208, 208);
  }

  // two rounds of block attention (in place on Bf)
  for (int r = 0; r < 2; ++r) {
    k_zero16<<<1, 32, 0, stream>>>(wsum);
    k_attn_tsum<<<1352, 256, 0, stream>>>(Bf, T);
    k_attn_dot<<<1352, 256, 0, stream>>>(Bf, T, wsum);
    long long total = 4LL * 104 * 104 * 128;
    k_attn_scale<<<dim3((unsigned)((total + 255) / 256)), 256, 0, stream>>>(Bf, wsum);
  }

  // decoder
  k_conv_wmma<128><<<dim3(85, 8, 4), 256, 0, stream>>>(Bf, dw1, db1, A, 104, 104, 128, 0, db1, db1, db1, db1);
  k_conv_wmma<128><<<dim3(85, 8, 4), 256, 0, stream>>>(A, dw2, db2, Bf, 104, 104, 128, 1, bn2g, bn2b, bn2m, bn2v);
  {
    long long total = 4LL * 208 * 208 * 128;
    k_up2x<<<dim3((unsigned)((total + 255) / 256)), 256, 0, stream>>>(Bf, A, 128, 104, 104);
  }
  k_conv_wmma<128><<<dim3(338, 4, 4), 256, 0, stream>>>(A, dw3, db3, Bf, 208, 208, 64, 0, db3, db3, db3, db3);
  k_conv_wmma<64><<<dim3(338, 4, 4), 256, 0, stream>>>(Bf, dw4, db4, A, 208, 208, 64, 1, bn4g, bn4b, bn4m, bn4v);
  {
    long long total = 4LL * 416 * 416 * 64;
    k_up2x<<<dim3((unsigned)((total + 255) / 256)), 256, 0, stream>>>(A, Bf, 64, 208, 208);
  }
  k_conv_last<<<dim3((unsigned)((4LL * 416 * 416 + 255) / 256)), 256, 0, stream>>>(Bf, dw5, db5, outp, 416, 416);
}
